// CausalSelfAttention_65962107732251
// MI455X (gfx1250) — matrix operand order repaired and checked by reading, not compiled, not run on hardware
//
#include <hip/hip_runtime.h>

#define T_SEQ 2048
#define EMB   2048
#define NH    16
#define NKV   4
#define HD    128

typedef __attribute__((ext_vector_type(16))) __bf16 v16bf;
typedef __attribute__((ext_vector_type(8)))  float  v8f;

union BFrag {
    v16bf v;
    uint4 q[2];
};

__device__ __forceinline__ unsigned short f2bf(float f) {
    unsigned int u = __builtin_bit_cast(unsigned int, f);
    unsigned int r = u + 0x7FFFu + ((u >> 16) & 1u);
    return (unsigned short)(r >> 16);
}
__device__ __forceinline__ float bf2f(unsigned short h) {
    unsigned int u = ((unsigned int)h) << 16;
    return __builtin_bit_cast(float, u);
}

// CDNA5 async global->LDS copy (16B per lane), tracked by ASYNCcnt.
// LDS address = low 32 bits of the generic pointer (LDS aperture truncation).
__device__ __forceinline__ void async_ld16(const void* gptr, void* lptr) {
    unsigned int lds      = (unsigned int)(size_t)lptr;
    unsigned long long ga = (unsigned long long)(size_t)gptr;
    asm volatile("global_load_async_to_lds_b128 %0, %1, off"
                 :
                 : "v"(lds), "v"(ga)
                 : "memory");
}
__device__ __forceinline__ void async_wait0() {
    asm volatile("s_wait_asynccnt 0x0" ::: "memory");
}

// ---------------------------------------------------------------- converts
__global__ void cvt_f32_bf16_kernel(const float* __restrict__ x,
                                    unsigned short* __restrict__ y, int n) {
    int i = blockIdx.x * 256 + threadIdx.x;
    if (i < n) y[i] = f2bf(x[i]);
}

// W[K][N] fp32 -> Wt[N][K] bf16 (tiled transpose through LDS)
__global__ __launch_bounds__(256)
void transpose_cvt_kernel(const float* __restrict__ W,
                          unsigned short* __restrict__ Wt, int K, int N) {
    __shared__ float tile[32][33];
    int tx = threadIdx.x & 31;
    int ty = threadIdx.x >> 5;           // 0..7
    int n0 = blockIdx.x * 32;
    int k0 = blockIdx.y * 32;
#pragma unroll
    for (int i = 0; i < 4; ++i)
        tile[ty * 4 + i][tx] = W[(size_t)(k0 + ty * 4 + i) * N + n0 + tx];
    __syncthreads();
#pragma unroll
    for (int i = 0; i < 4; ++i)
        Wt[(size_t)(n0 + ty * 4 + i) * K + k0 + tx] = f2bf(tile[tx][ty * 4 + i]);
}

// ---------------------------------------------------------------- GEMM
// C[M][N] = A[M][K](bf16) * Bt[N][K](bf16)^T ; fp32 accumulate.
// Block: 256 thr = 8 waves; tile 128x64; each wave 32x32 (2x2 wmma tiles).
// Double-buffered LDS, staged with global_load_async_to_lds_b128.
template <bool BF16OUT>
__global__ __launch_bounds__(256)
void gemm_bf16_kernel(const unsigned short* __restrict__ A,
                      const unsigned short* __restrict__ Bt,
                      void* __restrict__ Cout, int M, int N, int K) {
    __shared__ alignas(16) unsigned short As[2][128 * 32];
    __shared__ alignas(16) unsigned short Bs[2][64 * 32];
    const int tid  = threadIdx.x;
    const int lane = tid & 31;
    const int wave = tid >> 5;
    const int lh   = lane >> 4;   // half-group
    const int lm   = lane & 15;
    const int m0   = (wave >> 1) * 32;
    const int n0   = (wave & 1) * 32;
    const int blockM = blockIdx.y * 128;
    const int blockN = blockIdx.x * 64;

    // per-thread staging coordinates
    const int arow0 = (tid + 0)   >> 2, acol0 = ((tid + 0)   & 3) * 8;
    const int arow1 = (tid + 256) >> 2, acol1 = ((tid + 256) & 3) * 8;
    const int brow  = tid >> 2,         bcol  = (tid & 3) * 8;

    auto stage = [&](int buf, int kk) {
        async_ld16(&A[(size_t)(blockM + arow0) * K + kk + acol0],
                   &As[buf][arow0 * 32 + acol0]);
        async_ld16(&A[(size_t)(blockM + arow1) * K + kk + acol1],
                   &As[buf][arow1 * 32 + acol1]);
        async_ld16(&Bt[(size_t)(blockN + brow) * K + kk + bcol],
                   &Bs[buf][brow * 32 + bcol]);
    };

    v8f acc[2][2] = {};

    stage(0, 0);
    async_wait0();
    __syncthreads();

    int cur = 0;
    for (int kk = 0; kk < K; kk += 32) {
        if (kk + 32 < K) stage(cur ^ 1, kk + 32);

        BFrag af[2], bf[2];
#pragma unroll
        for (int mt = 0; mt < 2; ++mt) {
            int row = m0 + mt * 16 + lm;
            int c0  = lh * 8;
            af[mt].q[0] = *(const uint4*)&As[cur][row * 32 + c0];
            af[mt].q[1] = *(const uint4*)&As[cur][row * 32 + c0 + 16];
        }
#pragma unroll
        for (int nt = 0; nt < 2; ++nt) {
            int row = n0 + nt * 16 + lm;
            int c0  = lh * 8;
            bf[nt].q[0] = *(const uint4*)&Bs[cur][row * 32 + c0];
            bf[nt].q[1] = *(const uint4*)&Bs[cur][row * 32 + c0 + 16];
        }
#pragma unroll
        for (int mt = 0; mt < 2; ++mt)
#pragma unroll
            for (int nt = 0; nt < 2; ++nt)
                acc[mt][nt] = __builtin_amdgcn_wmma_f32_16x16x32_bf16(
                    false, af[mt].v, false, bf[nt].v, (short)0, acc[mt][nt],
                    false, false);

        async_wait0();
        __syncthreads();
        cur ^= 1;
    }

#pragma unroll
    for (int mt = 0; mt < 2; ++mt)
#pragma unroll
        for (int nt = 0; nt < 2; ++nt)
#pragma unroll
            for (int r = 0; r < 8; ++r) {
                int m = blockM + m0 + mt * 16 + r + lh * 8;
                int n = blockN + n0 + nt * 16 + lm;
                float v = acc[mt][nt][r];
                if (BF16OUT)
                    ((unsigned short*)Cout)[(size_t)m * N + n] = f2bf(v);
                else
                    ((float*)Cout)[(size_t)m * N + n] = v;
            }
}

// ---------------------------------------------------------------- RoPE
// buf: [B*T][nH*128] bf16, in-place. Matches reference repeat-2 freq layout.
__global__ void rope_kernel(unsigned short* __restrict__ buf, int nH,
                            const int* __restrict__ posp, int total) {
    int idx = blockIdx.x * 256 + threadIdx.x;
    if (idx >= total) return;
    int j   = idx & 63;
    int hh  = (idx >> 6) % nH;
    int row = idx / (64 * nH);
    float t = (float)(posp[0] + (row & (T_SEQ - 1)));
    size_t base = (size_t)row * (nH * HD) + hh * HD;
    float a = bf2f(buf[base + j]);
    float b = bf2f(buf[base + j + 64]);
    const float lg = 9.210340371976184f;  // ln(10000)
    int   i1 = j >> 1;
    float f1 = t * __expf(-lg * (float)i1 * (1.0f / 64.0f));
    float f2 = t * __expf(-lg * (float)(32 + i1) * (1.0f / 64.0f));
    buf[base + j]      = f2bf(a * cosf(f1) - b * sinf(f1));
    buf[base + j + 64] = f2bf(b * cosf(f2) + a * sinf(f2));
}

// ---------------------------------------------------------------- attention
// 128 thr = 4 waves per block; block owns 64 q-rows of one (b,h).
__global__ __launch_bounds__(128)
void attn_kernel(const unsigned short* __restrict__ Qb,
                 const unsigned short* __restrict__ Kb,
                 const unsigned short* __restrict__ Vb,
                 unsigned short* __restrict__ Ob) {
    __shared__ alignas(16) unsigned short Ks[32 * 128];   // [key][d]
    __shared__ alignas(16) unsigned short Vs[128 * 32];   // [d][key]
    __shared__ alignas(16) unsigned short Ps[4][16 * 32]; // per-wave P scratch

    const int tid  = threadIdx.x;
    const int lane = tid & 31;
    const int wave = tid >> 5;
    const int lh   = lane >> 4;
    const int lm   = lane & 15;

    const int QT = T_SEQ / 64;
    int qt = blockIdx.x % QT;
    int bh = blockIdx.x / QT;
    int h  = bh % NH;
    int b  = bh / NH;
    int kh = h >> 2;
    const int t0 = qt * 64 + wave * 16;

    // preload Q fragments for this wave's 16 rows (D=128 -> 4 k-steps)
    BFrag qf[4];
    {
        size_t qrow = ((size_t)(b * T_SEQ) + t0 + lm) * EMB + h * HD;
#pragma unroll
        for (int c = 0; c < 4; ++c) {
            qf[c].q[0] = *(const uint4*)&Qb[qrow + c * 32 + lh * 8];
            qf[c].q[1] = *(const uint4*)&Qb[qrow + c * 32 + lh * 8 + 16];
        }
    }

    v8f   o[8] = {};
    float mrow[8], lrow[8];
#pragma unroll
    for (int r = 0; r < 8; ++r) { mrow[r] = -1e30f; lrow[r] = 0.0f; }
    const float scale = 0.08838834764831845f;  // 1/sqrt(128)

    const int nKB = qt * 2 + 2;
    for (int kb = 0; kb < nKB; ++kb) {
        const int s0 = kb * 32;
        // stage K tile [32][128] via async global->LDS
#pragma unroll
        for (int i = 0; i < 4; ++i) {
            int e = tid + i * 128;
            int row = e >> 4;
            int col = (e & 15) * 8;
            async_ld16(&Kb[((size_t)(b * T_SEQ) + s0 + row) * (NKV * HD) +
                           kh * HD + col],
                       &Ks[row * 128 + col]);
        }
        // stage V tile transposed -> Vs[d][key]  (byte scatter, stays manual)
#pragma unroll
        for (int i = 0; i < 4; ++i) {
            int e = tid + i * 128;
            int row = e >> 4;
            int col = (e & 15) * 8;
            uint4 d4 = *(const uint4*)&Vb[((size_t)(b * T_SEQ) + s0 + row) *
                                              (NKV * HD) + kh * HD + col];
            unsigned int w[4] = {d4.x, d4.y, d4.z, d4.w};
#pragma unroll
            for (int jj = 0; jj < 4; ++jj) {
                Vs[(col + 2 * jj) * 32 + row]     = (unsigned short)(w[jj] & 0xFFFF);
                Vs[(col + 2 * jj + 1) * 32 + row] = (unsigned short)(w[jj] >> 16);
            }
        }
        async_wait0();
        __syncthreads();

        // S = Q K^T : two 16-col tiles, 4 k-steps over D
        v8f s[2] = {};
#pragma unroll
        for (int nt = 0; nt < 2; ++nt) {
            int krow = (nt * 16 + lm) * 128;
#pragma unroll
            for (int c = 0; c < 4; ++c) {
                BFrag kf;
                int c0 = c * 32 + lh * 8;
                kf.q[0] = *(const uint4*)&Ks[krow + c0];
                kf.q[1] = *(const uint4*)&Ks[krow + c0 + 16];
                s[nt] = __builtin_amdgcn_wmma_f32_16x16x32_bf16(
                    false, qf[c].v, false, kf.v, (short)0, s[nt], false, false);
            }
        }

        // scale + causal mask
        float sv[2][8];
#pragma unroll
        for (int nt = 0; nt < 2; ++nt)
#pragma unroll
            for (int r = 0; r < 8; ++r) {
                int si = s0 + nt * 16 + lm;
                int ti = t0 + r + lh * 8;
                float v = s[nt][r] * scale;
                sv[nt][r] = (si > ti) ? -1e30f : v;
            }

        // online softmax, write P (bf16) into per-wave LDS scratch
        float corr[8];
#pragma unroll
        for (int r = 0; r < 8; ++r) {
            float mx = fmaxf(sv[0][r], sv[1][r]);
#pragma unroll
            for (int off = 8; off >= 1; off >>= 1)
                mx = fmaxf(mx, __shfl_xor(mx, off, 32));
            float mnew = fmaxf(mrow[r], mx);
            corr[r] = __expf(mrow[r] - mnew);
            mrow[r] = mnew;
            float p0 = __expf(sv[0][r] - mnew);
            float p1 = __expf(sv[1][r] - mnew);
            float rs = p0 + p1;
#pragma unroll
            for (int off = 8; off >= 1; off >>= 1)
                rs += __shfl_xor(rs, off, 32);
            lrow[r] = lrow[r] * corr[r] + rs;
            int prow = (r + lh * 8) * 32;
            Ps[wave][prow + lm]      = f2bf(p0);
            Ps[wave][prow + 16 + lm] = f2bf(p1);
        }
#pragma unroll
        for (int dt = 0; dt < 8; ++dt)
#pragma unroll
            for (int r = 0; r < 8; ++r) o[dt][r] *= corr[r];

        // reload P as an A-fragment (LDS ops from same wave are in-order)
        BFrag pf;
        pf.q[0] = *(const uint4*)&Ps[wave][lm * 32 + lh * 8];
        pf.q[1] = *(const uint4*)&Ps[wave][lm * 32 + lh * 8 + 16];

        // O += P V : 8 d-tiles, one 32-deep k-step
#pragma unroll
        for (int dt = 0; dt < 8; ++dt) {
            BFrag vf;
            int vrow = (dt * 16 + lm) * 32;
            vf.q[0] = *(const uint4*)&Vs[vrow + lh * 8];
            vf.q[1] = *(const uint4*)&Vs[vrow + lh * 8 + 16];
            o[dt] = __builtin_amdgcn_wmma_f32_16x16x32_bf16(
                false, pf.v, false, vf.v, (short)0, o[dt], false, false);
        }
        __syncthreads();
    }

    // normalize and write bf16 attention output [B*T][E]
#pragma unroll
    for (int r = 0; r < 8; ++r) {
        float inv = 1.0f / lrow[r];
        int t = t0 + r + lh * 8;
        size_t orow = ((size_t)(b * T_SEQ) + t) * EMB + h * HD;
#pragma unroll
        for (int dt = 0; dt < 8; ++dt)
            Ob[orow + dt * 16 + lm] = f2bf(o[dt][r] * inv);
    }
}

// ---------------------------------------------------------------- launch
extern "C" void kernel_launch(void* const* d_in, const int* in_sizes, int n_in,
                              void* d_out, int out_size, void* d_ws,
                              size_t ws_size, hipStream_t stream) {
    const float* x  = (const float*)d_in[0];
    const float* Wq = (const float*)d_in[1];
    const float* Wk = (const float*)d_in[2];
    const float* Wv = (const float*)d_in[3];
    const float* Wp = (const float*)d_in[4];
    const int*  pos = (const int*)d_in[5];

    const int M  = 2 * T_SEQ;        // 4096
    const int KV = NKV * HD;         // 512

    char* ws = (char*)d_ws;
    size_t off = 0;
    auto alloc = [&](size_t bytes) {
        void* p = ws + off;
        off += (bytes + 255) & ~(size_t)255;
        return p;
    };
    unsigned short* xb  = (unsigned short*)alloc((size_t)M * EMB * 2);
    unsigned short* Wqt = (unsigned short*)alloc((size_t)EMB * EMB * 2);
    unsigned short* Wkt = (unsigned short*)alloc((size_t)KV * EMB * 2);
    unsigned short* Wvt = (unsigned short*)alloc((size_t)KV * EMB * 2);
    unsigned short* Wpt = (unsigned short*)alloc((size_t)EMB * EMB * 2);
    unsigned short* Qb  = (unsigned short*)alloc((size_t)M * EMB * 2);
    unsigned short* Kb  = (unsigned short*)alloc((size_t)M * KV * 2);
    unsigned short* Vb  = (unsigned short*)alloc((size_t)M * KV * 2);
    unsigned short* Ab  = (unsigned short*)alloc((size_t)M * EMB * 2);

    // x -> bf16
    {
        int n = M * EMB;
        cvt_f32_bf16_kernel<<<(n + 255) / 256, 256, 0, stream>>>(x, xb, n);
    }
    // weights -> transposed bf16 [N][K]
    transpose_cvt_kernel<<<dim3(EMB / 32, EMB / 32), 256, 0, stream>>>(Wq, Wqt, EMB, EMB);
    transpose_cvt_kernel<<<dim3(KV / 32, EMB / 32), 256, 0, stream>>>(Wk, Wkt, EMB, KV);
    transpose_cvt_kernel<<<dim3(KV / 32, EMB / 32), 256, 0, stream>>>(Wv, Wvt, EMB, KV);
    transpose_cvt_kernel<<<dim3(EMB / 32, EMB / 32), 256, 0, stream>>>(Wp, Wpt, EMB, EMB);

    // QKV projections
    gemm_bf16_kernel<true><<<dim3(EMB / 64, M / 128), 256, 0, stream>>>(
        xb, Wqt, Qb, M, EMB, EMB);
    gemm_bf16_kernel<true><<<dim3(KV / 64, M / 128), 256, 0, stream>>>(
        xb, Wkt, Kb, M, KV, EMB);
    gemm_bf16_kernel<true><<<dim3(KV / 64, M / 128), 256, 0, stream>>>(
        xb, Wvt, Vb, M, KV, EMB);

    // RoPE on Q and K
    {
        int nq = M * NH * 64;
        rope_kernel<<<(nq + 255) / 256, 256, 0, stream>>>(Qb, NH, pos, nq);
        int nk = M * NKV * 64;
        rope_kernel<<<(nk + 255) / 256, 256, 0, stream>>>(Kb, NKV, pos, nk);
    }

    // flash attention
    attn_kernel<<<2 * NH * (T_SEQ / 64), 128, 0, stream>>>(Qb, Kb, Vb, Ab);

    // output projection -> fp32 d_out
    gemm_bf16_kernel<false><<<dim3(EMB / 64, M / 128), 256, 0, stream>>>(
        Ab, Wpt, d_out, M, EMB, EMB);
}
